// MultiHeadSeqAttention_58686433133147
// MI455X (gfx1250) — matrix operand order repaired and checked by reading, not compiled, not run on hardware
//
#include <hip/hip_runtime.h>
#include <hip/hip_bf16.h>

// ---------------------------------------------------------------------------
// MultiHeadSeqAttention (Transformer-XL style banded attention) for gfx1250.
// B=8, M=512, L=1024, H=1024, NH=16, D=64, T=M+L=1536, BK=B*NH=128.
// Pipeline:
//   1) proj_heads_kernel x3 : q/k/v = x @ W^T (f32 in, f16 out). v is written
//      transposed [bk][d][t] (padded t) so attn B-fragments are contiguous.
//   2) pe_transpose_kernel  : key_pe (D,L) f32 -> (L,D) f16, + vT pad zeroing
//   3) attn_kernel          : flash-style banded attention, WMMA f16
//   4) out_proj_kernel      : out = o @ Wo^T  (f16 in, f32 out), WMMA f16
// ---------------------------------------------------------------------------

typedef __attribute__((ext_vector_type(16))) _Float16 v16h;
typedef __attribute__((ext_vector_type(8)))  _Float16 v8h;
typedef __attribute__((ext_vector_type(4)))  _Float16 v4h;
typedef __attribute__((ext_vector_type(8)))  float    v8f;

union F16x16 { v16h v; _Float16 h[16]; };
union F16x8  { v8h  v; _Float16 h[8];  };
union F16x4  { v4h  v; _Float16 h[4];  };
union F32x8  { v8f  v; float    f[8];  };

#define HDIM 1024
#define NH   16
#define DDIM 64
#define MQ   512
#define LWIN 1024
#define TKV  1536
#define TPAD 1568   // TKV + 32 pad so attn v-loads never need clamping
#define BKH  128

static __device__ __forceinline__ v8f wmma_f16(v16h a, v16h b, v8f c) {
  // D = A(16x32 f16) * B(32x16 f16) + C(16x16 f32)
  return __builtin_amdgcn_wmma_f32_16x16x32_f16(false, a, false, b, (short)0, c,
                                                false, false);
}

// A-fragment element i -> K index (16-bit A 16x32 layout, g = lane>>4)
static __device__ __forceinline__ int a_kidx(int i, int g) {
  return (i < 8) ? (g * 8 + i) : (16 + g * 8 + (i - 8));
}

// ---------------------------------------------------------------------------
// 1) Projection GEMM: Y[n,h] = sum_c X[n,c] * W[h,c], f16 head-major out.
//    ROWS in {512,1536} is a template constant (no runtime division).
//    VT=false: Y[((b*NH+head)*ROWS + t)*64 + d]
//    VT=true : Y[((b*NH+head)*64 + d)*TPAD + t]   (transposed, padded)
//    Block tile 128x64 (one head), 8 waves, K-step 32.
// ---------------------------------------------------------------------------
template <int ROWS, bool VT>
__global__ __launch_bounds__(256) void proj_heads_kernel(
    const float* __restrict__ X, const float* __restrict__ W,
    _Float16* __restrict__ Y)
{
  __shared__ _Float16 As[128][40];
  __shared__ _Float16 Bs[64][40];
  const int tid  = threadIdx.x;
  const int wid  = tid >> 5;
  const int lane = tid & 31;
  const int g    = lane >> 4;
  const int ln   = lane & 15;
  const int n0   = blockIdx.x * 128;
  const int head = blockIdx.y;
  // whole 128-row tile lies in one batch (ROWS % 128 == 0)
  const int bIdx   = n0 / ROWS;
  const int trBase = n0 - bIdx * ROWS;

  F32x8 acc[4];
  #pragma unroll
  for (int t = 0; t < 4; ++t)
    #pragma unroll
    for (int j = 0; j < 8; ++j) acc[t].f[j] = 0.f;

  for (int k0 = 0; k0 < HDIM; k0 += 32) {
    // --- batched loads into registers, then convert+store to LDS ---
    float4 xa[4];
    #pragma unroll
    for (int i = 0; i < 4; ++i) {               // A: 128x32 f32 (1024 float4)
      int idx = tid + 256 * i;
      int r = idx >> 3, c = (idx & 7) * 4;
      xa[i] = *(const float4*)&X[(size_t)(n0 + r) * HDIM + k0 + c];
    }
    float4 xb[2];
    #pragma unroll
    for (int i = 0; i < 2; ++i) {               // B: 64x32 f32 (512 float4)
      int idx = tid + 256 * i;
      int r = idx >> 3, c = (idx & 7) * 4;
      xb[i] = *(const float4*)&W[(size_t)(head * 64 + r) * HDIM + k0 + c];
    }
    #pragma unroll
    for (int i = 0; i < 4; ++i) {
      int idx = tid + 256 * i;
      int r = idx >> 3, c = (idx & 7) * 4;
      F16x4 u;
      u.h[0] = (_Float16)xa[i].x; u.h[1] = (_Float16)xa[i].y;
      u.h[2] = (_Float16)xa[i].z; u.h[3] = (_Float16)xa[i].w;
      *(v4h*)&As[r][c] = u.v;
    }
    #pragma unroll
    for (int i = 0; i < 2; ++i) {
      int idx = tid + 256 * i;
      int r = idx >> 3, c = (idx & 7) * 4;
      F16x4 u;
      u.h[0] = (_Float16)xb[i].x; u.h[1] = (_Float16)xb[i].y;
      u.h[2] = (_Float16)xb[i].z; u.h[3] = (_Float16)xb[i].w;
      *(v4h*)&Bs[r][c] = u.v;
    }
    __syncthreads();

    F16x16 a;
    const int arow = 16 * wid + ln;
    #pragma unroll
    for (int i = 0; i < 16; ++i) a.h[i] = As[arow][a_kidx(i, g)];

    #pragma unroll
    for (int t = 0; t < 4; ++t) {
      F16x16 bf;
      #pragma unroll
      for (int i = 0; i < 16; ++i) bf.h[i] = Bs[t * 16 + ln][a_kidx(i, g)];
      acc[t].v = wmma_f16(a.v, bf.v, acc[t].v);
    }
    __syncthreads();
  }

  const size_t bkBase = (size_t)(bIdx * NH + head);
  #pragma unroll
  for (int t = 0; t < 4; ++t)
    #pragma unroll
    for (int j = 0; j < 8; ++j) {
      const int tr = trBase + 16 * wid + j + 8 * g;
      const int d  = t * 16 + ln;
      const _Float16 val = (_Float16)acc[t].f[j];
      if (VT)
        Y[(bkBase * DDIM + d) * TPAD + tr] = val;
      else
        Y[(bkBase * ROWS + tr) * DDIM + d] = val;
    }
}

// ---------------------------------------------------------------------------
// 2a) key_pe (1,D,L) f32 -> peT (L,D) f16
// ---------------------------------------------------------------------------
__global__ __launch_bounds__(256) void pe_transpose_kernel(
    const float* __restrict__ pe, _Float16* __restrict__ peT)
{
  int idx = blockIdx.x * 256 + threadIdx.x;     // 0 .. 65535
  int d = idx >> 10, l = idx & 1023;
  peT[(size_t)l * DDIM + d] = (_Float16)pe[idx];
}

// 2b) zero the t-pad of vT: t in [TKV, TPAD) for all 128*64 rows
__global__ __launch_bounds__(256) void vpad_fill_kernel(_Float16* __restrict__ vT)
{
  int idx = blockIdx.x * 256 + threadIdx.x;     // 0 .. 128*64*32-1
  int row = idx >> 5, tp = TKV + (idx & 31);
  vT[(size_t)row * TPAD + tp] = (_Float16)0.f;
}

// ---------------------------------------------------------------------------
// 3) Banded flash attention. Grid: 128*8 blocks, 128 threads (4 waves).
//    Each wave owns one 16-row query strip m0..m0+15 of head bk.
//    Band: query m attends keys t in [m, m+L). Rect coords: c = t - m0,
//    in-band iff r <= c < r + L.  score = (q.k + P[r][c-r]) / 8.
// ---------------------------------------------------------------------------
#define ATTN_SMEM_BYTES ((4 * 16 * 1024 + 4 * 16 * 32) * 2)

__global__ __launch_bounds__(128) void attn_kernel(
    const _Float16* __restrict__ qh, const _Float16* __restrict__ kh,
    const _Float16* __restrict__ vT, const _Float16* __restrict__ peT,
    _Float16* __restrict__ oh)
{
  extern __shared__ char smem[];
  const int tid  = threadIdx.x;
  const int wid  = tid >> 5;
  const int lane = tid & 31;
  const int g    = lane >> 4;
  const int ln   = lane & 15;
  const int bk   = blockIdx.x >> 3;             // 0..127
  const int sgrp = blockIdx.x & 7;              // 0..7
  const int m0   = (sgrp * 4 + wid) * 16;       // strip base query row
  const int b    = bk >> 4;
  const int head = bk & 15;

  _Float16* Pw    = (_Float16*)smem + (size_t)wid * (16 * 1024);  // 32KB/wave
  _Float16* stage = (_Float16*)smem + 4 * (16 * 1024) + wid * (16 * 32);

  // q A-fragments (K = d 0..31 and 32..63), resident for whole kernel
  F16x16 qa0, qa1;
  {
    const size_t qrow = ((size_t)bk * MQ + m0 + ln) * DDIM;
    #pragma unroll
    for (int i = 0; i < 16; ++i) {
      int kk = a_kidx(i, g);
      qa0.h[i] = qh[qrow + kk];
      qa1.h[i] = qh[qrow + kk + 32];
    }
  }

  // ---- Phase 1: P[r][l] = q_r . pe[:,l]   (16 x 1024, f16 in LDS) ----
  for (int lt = 0; lt < 64; ++lt) {
    const int l0 = lt * 16;
    F16x16 pb0, pb1;
    const size_t pcol = (size_t)(l0 + ln) * DDIM;
    #pragma unroll
    for (int i = 0; i < 16; ++i) {
      pb0.h[i] = peT[pcol + a_kidx(i, g)];
      pb1.h[i] = peT[pcol + a_kidx(i, g) + 32];
    }
    F32x8 pacc;
    #pragma unroll
    for (int j = 0; j < 8; ++j) pacc.f[j] = 0.f;
    pacc.v = wmma_f16(qa0.v, pb0.v, pacc.v);
    pacc.v = wmma_f16(qa1.v, pb1.v, pacc.v);
    #pragma unroll
    for (int j = 0; j < 8; ++j)
      Pw[(j + 8 * g) * 1024 + l0 + ln] = (_Float16)pacc.f[j];
  }
  asm volatile("s_wait_dscnt 0" ::: "memory");  // same-wave LDS RAW

  // ---- Phase 2: stream 32-key tiles with online softmax ----
  float mrow[8], lrow[8];
  F32x8 O[4];
  #pragma unroll
  for (int j = 0; j < 8; ++j) { mrow[j] = -3.0e38f; lrow[j] = 0.f; }
  #pragma unroll
  for (int t = 0; t < 4; ++t)
    #pragma unroll
    for (int j = 0; j < 8; ++j) O[t].f[j] = 0.f;

  for (int step = 0; step < 33; ++step) {       // 33*32 = 1056 >= L+15
    const int c0 = step * 32;
    // prefetch next key tile into caches (global_prefetch_b8)
    {
      int tp = m0 + c0 + 32 + ln;
      tp = min(tp, TKV - 1);
      __builtin_prefetch((const void*)(kh + ((size_t)bk * TKV + tp) * DDIM), 0, 3);
    }
    float sv[2][8];
    #pragma unroll
    for (int sub = 0; sub < 2; ++sub) {
      const int cc0 = c0 + sub * 16;
      int t = m0 + cc0 + ln;
      t = min(t, TKV - 1);
      const size_t krow = ((size_t)bk * TKV + t) * DDIM;
      F16x16 kb0, kb1;
      #pragma unroll
      for (int i = 0; i < 16; ++i) {
        kb0.h[i] = kh[krow + a_kidx(i, g)];
        kb1.h[i] = kh[krow + a_kidx(i, g) + 32];
      }
      F32x8 s;
      #pragma unroll
      for (int j = 0; j < 8; ++j) s.f[j] = 0.f;
      s.v = wmma_f16(qa0.v, kb0.v, s.v);
      s.v = wmma_f16(qa1.v, kb1.v, s.v);
      #pragma unroll
      for (int j = 0; j < 8; ++j) {
        const int r = j + 8 * g;
        const int c = cc0 + ln;
        const bool band = (c >= r) && (c - r < LWIN);
        const float pv = band ? (float)Pw[r * 1024 + (c - r)] : 0.f;
        sv[sub][j] = band ? (s.f[j] + pv) * 0.125f : -3.0e38f;
      }
    }
    // per-row max over the 32 columns, online-softmax update
    float fsc[8], psum[8];
    #pragma unroll
    for (int j = 0; j < 8; ++j) {
      float mx = fmaxf(sv[0][j], sv[1][j]);
      mx = fmaxf(mx, __shfl_xor(mx, 8));
      mx = fmaxf(mx, __shfl_xor(mx, 4));
      mx = fmaxf(mx, __shfl_xor(mx, 2));
      mx = fmaxf(mx, __shfl_xor(mx, 1));
      const float mnew = fmaxf(mrow[j], mx);
      fsc[j] = __expf(mrow[j] - mnew);
      mrow[j] = mnew;
      const float p0 = __expf(sv[0][j] - mnew);
      const float p1 = __expf(sv[1][j] - mnew);
      stage[(j + 8 * g) * 32 + ln]      = (_Float16)p0;
      stage[(j + 8 * g) * 32 + 16 + ln] = (_Float16)p1;
      float ps = p0 + p1;
      ps += __shfl_xor(ps, 8);
      ps += __shfl_xor(ps, 4);
      ps += __shfl_xor(ps, 2);
      ps += __shfl_xor(ps, 1);
      psum[j] = ps;
    }
    #pragma unroll
    for (int j = 0; j < 8; ++j) lrow[j] = lrow[j] * fsc[j] + psum[j];

    asm volatile("s_wait_dscnt 0" ::: "memory");
    F16x16 pa;                                  // C-layout -> A-layout restage
    #pragma unroll
    for (int i = 0; i < 16; ++i) pa.h[i] = stage[ln * 32 + 16 * g + i];

    #pragma unroll
    for (int t = 0; t < 4; ++t)
      #pragma unroll
      for (int j = 0; j < 8; ++j) O[t].f[j] *= fsc[j];

    #pragma unroll
    for (int dt = 0; dt < 4; ++dt) {
      // vT[bk][d][t]: 16 contiguous halves per lane (pad makes clamp unneeded)
      const size_t vrow =
          ((size_t)bk * DDIM + dt * 16 + ln) * TPAD + (m0 + c0 + 16 * g);
      F16x16 vb;
      #pragma unroll
      for (int i = 0; i < 16; ++i) vb.h[i] = vT[vrow + i];
      O[dt].v = wmma_f16(pa.v, vb.v, O[dt].v);
    }
  }

  // finalize: O /= rowsum, write f16 to (B,M,H) with h = head*64 + d
  #pragma unroll
  for (int j = 0; j < 8; ++j) lrow[j] = 1.f / lrow[j];
  #pragma unroll
  for (int t = 0; t < 4; ++t)
    #pragma unroll
    for (int j = 0; j < 8; ++j) {
      const int m = m0 + j + 8 * g;
      oh[((size_t)b * MQ + m) * HDIM + head * 64 + t * 16 + ln] =
          (_Float16)(O[t].f[j] * lrow[j]);
    }
}

// ---------------------------------------------------------------------------
// 4) Output projection: out[n,h] = sum_c oh[n,c] * Wo[h,c], f32 out.
// ---------------------------------------------------------------------------
__global__ __launch_bounds__(256) void out_proj_kernel(
    const _Float16* __restrict__ A, const float* __restrict__ W,
    float* __restrict__ Y)
{
  __shared__ _Float16 As[128][40];
  __shared__ _Float16 Bs[64][40];
  const int tid  = threadIdx.x;
  const int wid  = tid >> 5;
  const int lane = tid & 31;
  const int g    = lane >> 4;
  const int ln   = lane & 15;
  const int n0   = blockIdx.x * 128;
  const int h0   = blockIdx.y * 64;

  F32x8 acc[4];
  #pragma unroll
  for (int t = 0; t < 4; ++t)
    #pragma unroll
    for (int j = 0; j < 8; ++j) acc[t].f[j] = 0.f;

  for (int k0 = 0; k0 < HDIM; k0 += 32) {
    v8h va[2];
    #pragma unroll
    for (int i = 0; i < 2; ++i) {               // A: 128x32 f16 (512 v8h)
      int idx = tid + 256 * i;
      int r = idx >> 2, c = (idx & 3) * 8;
      va[i] = *(const v8h*)&A[(size_t)(n0 + r) * HDIM + k0 + c];
    }
    float4 xb[2];
    #pragma unroll
    for (int i = 0; i < 2; ++i) {               // B: 64x32 f32 (512 float4)
      int idx = tid + 256 * i;
      int r = idx >> 3, c = (idx & 7) * 4;
      xb[i] = *(const float4*)&W[(size_t)(h0 + r) * HDIM + k0 + c];
    }
    #pragma unroll
    for (int i = 0; i < 2; ++i) {
      int idx = tid + 256 * i;
      int r = idx >> 2, c = (idx & 3) * 8;
      *(v8h*)&As[r][c] = va[i];
    }
    #pragma unroll
    for (int i = 0; i < 2; ++i) {
      int idx = tid + 256 * i;
      int r = idx >> 3, c = (idx & 7) * 4;
      F16x4 u;
      u.h[0] = (_Float16)xb[i].x; u.h[1] = (_Float16)xb[i].y;
      u.h[2] = (_Float16)xb[i].z; u.h[3] = (_Float16)xb[i].w;
      *(v4h*)&Bs[r][c] = u.v;
    }
    __syncthreads();

    F16x16 a;
    const int arow = 16 * wid + ln;
    #pragma unroll
    for (int i = 0; i < 16; ++i) a.h[i] = As[arow][a_kidx(i, g)];

    #pragma unroll
    for (int t = 0; t < 4; ++t) {
      F16x16 bf;
      #pragma unroll
      for (int i = 0; i < 16; ++i) bf.h[i] = Bs[t * 16 + ln][a_kidx(i, g)];
      acc[t].v = wmma_f16(a.v, bf.v, acc[t].v);
    }
    __syncthreads();
  }

  #pragma unroll
  for (int t = 0; t < 4; ++t)
    #pragma unroll
    for (int j = 0; j < 8; ++j) {
      const int grow = n0 + 16 * wid + j + 8 * g;
      Y[(size_t)grow * HDIM + h0 + t * 16 + ln] = acc[t].f[j];
    }
}

// ---------------------------------------------------------------------------
// Host launcher
// ---------------------------------------------------------------------------
extern "C" void kernel_launch(void* const* d_in, const int* in_sizes, int n_in,
                              void* d_out, int out_size, void* d_ws, size_t ws_size,
                              hipStream_t stream) {
  (void)in_sizes; (void)n_in; (void)out_size; (void)ws_size;
  const float* query  = (const float*)d_in[0];   // (8,512,1024)
  const float* key    = (const float*)d_in[1];   // (8,1536,1024)
  const float* value  = (const float*)d_in[2];   // (8,1536,1024)
  const float* key_pe = (const float*)d_in[3];   // (1,64,1024)
  const float* Wq     = (const float*)d_in[4];
  const float* Wk     = (const float*)d_in[5];
  const float* Wv     = (const float*)d_in[6];
  const float* Wo     = (const float*)d_in[7];
  float* out = (float*)d_out;                    // (8,512,1024)

  char* ws = (char*)d_ws;
  constexpr size_t QH_BYTES = (size_t)BKH * MQ  * DDIM * 2;   //  8 MB
  constexpr size_t KH_BYTES = (size_t)BKH * TKV * DDIM * 2;   // 24 MB
  constexpr size_t VT_BYTES = (size_t)BKH * DDIM * TPAD * 2;  // 24.5 MB
  constexpr size_t PE_BYTES = (size_t)LWIN * DDIM * 2;        // 128 KB
  _Float16* qh  = (_Float16*)(ws);
  _Float16* kh  = (_Float16*)(ws + QH_BYTES);
  _Float16* vT  = (_Float16*)(ws + QH_BYTES + KH_BYTES);
  _Float16* peT = (_Float16*)(ws + QH_BYTES + KH_BYTES + VT_BYTES);
  _Float16* oh  = (_Float16*)(ws + QH_BYTES + KH_BYTES + VT_BYTES + PE_BYTES);

  // 1) projections into f16 (v transposed + padded)
  proj_heads_kernel<MQ,  false><<<dim3(32, 16), 256, 0, stream>>>(query, Wq, qh);
  proj_heads_kernel<TKV, false><<<dim3(96, 16), 256, 0, stream>>>(key,   Wk, kh);
  proj_heads_kernel<TKV, true ><<<dim3(96, 16), 256, 0, stream>>>(value, Wv, vT);
  // 2) positional table transpose/convert + vT pad zero
  pe_transpose_kernel<<<256, 256, 0, stream>>>(key_pe, peT);
  vpad_fill_kernel<<<1024, 256, 0, stream>>>(vT);
  // 3) banded flash attention (4 waves/block, 132KB dynamic LDS)
  attn_kernel<<<BKH * 8, 128, ATTN_SMEM_BYTES, stream>>>(qh, kh, vT, peT, oh);
  // 4) output projection
  out_proj_kernel<<<dim3(32, 16), 256, 0, stream>>>(oh, Wo, out);
}
